// SemanticMapSelfAttentionLSTMModel_66151086293051
// MI455X (gfx1250) — matrix operand order repaired and checked by reading, compiled, not run on hardware
//
#include <hip/hip_runtime.h>
#include <math.h>

// ---------------------------------------------------------------------------
// Types for CDNA5 WMMA
// ---------------------------------------------------------------------------
typedef __attribute__((ext_vector_type(16))) _Float16 v16h;
typedef __attribute__((ext_vector_type(8)))  _Float16 v8h;
typedef __attribute__((ext_vector_type(8)))  float    v8f;

#define CONV_MT 64   // output-channel tile (4 waves x 16)
#define CONV_NT 64   // pixel tile (4 sub-tiles of 16 per wave)
#define CONV_KT 32   // K step (matches v_wmma_f32_16x16x32_f16)

// ---------------------------------------------------------------------------
// Implicit-GEMM convolution with fused BN (scale/bias), optional residual add
// and ReLU.  A = weights (Cout x Kg), B = im2col activations (Kg x pixels).
// One 128-thread block computes a 64x64 output tile.  Each wave owns one A
// fragment (16 rows) and 4 accumulators; 4 v_wmma_f32_16x16x32_f16 per K step.
//
// LDS tiles are stored pre-swizzled in this kernel's contiguous-k layout (same for A and B) so a
// lane's fragment is one contiguous 32-byte chunk (2x ds_load_b128):
//   A (16-bit A-matrix 16x32): elem(cK) = cK&15,
//                              lane     = (cK&16) + (m&15)
//   B (16-bit B-matrix 32x16): elem = kr&15, lane = (kr&16) + nc
// All staging gathers are branchless (clamped address + cndmask select) so
// global loads issue back-to-back with a single wait.
// ---------------------------------------------------------------------------
__global__ __launch_bounds__(128)
void conv_bn_wmma_kernel(const float* __restrict__ x,
                         const float* __restrict__ w,       // OIHW == [Cout][Kg]
                         const float* __restrict__ gamma,
                         const float* __restrict__ beta,
                         const float* __restrict__ residual, // may be null
                         float* __restrict__ out,
                         int N, int Cin, int Hin, int Win,
                         int Cout, int KH, int KW, int stride, int pad,
                         int Hout, int Wout, int relu_flag)
{
    __shared__ __align__(32) _Float16 Asw[4][32][16];  // [wave][lane][elem]
    __shared__ __align__(32) _Float16 Bsw[4][32][16];  // [subtile][lane][elem]

    const int tid  = threadIdx.x;
    const int lane = tid & 31;
    const int wave = tid >> 5;

    const int Kg        = Cin * KH * KW;
    const int KHW       = KH * KW;
    const int HW        = Hout * Wout;
    const int tot_pix   = N * HW;
    const int npix_base = blockIdx.x * CONV_NT;
    const int oc_base   = blockIdx.y * CONV_MT;

    // ---- hoisted per-thread staging geometry ----
    // A staging: thread -> (row r, k-half kbsel)
    const int  ar     = tid >> 1;               // 0..63 local output channel
    const int  kbsel  = (tid & 1) << 4;         // 0 or 16
    const int  aoc    = oc_base + ar;
    const bool aocv   = (aoc < Cout);
    const float* wrow = w + (size_t)(aocv ? aoc : 0) * Kg;
    _Float16* adst_lo = &Asw[ar >> 4][kbsel + (ar & 15)][0];
    _Float16* adst_hi = &Asw[ar >> 4][kbsel + (ar & 15)][8];

    // B staging: thread -> (pixel column gnc, k-row block krb)
    const int  gnc  = tid & 63;
    const int  krb  = (tid >> 6) << 4;          // 0 or 16
    const int  gp   = npix_base + gnc;
    const bool gpv  = (gp < tot_pix);
    {
        // nothing
    }
    const int  gpc  = gpv ? gp : 0;
    const int  bn   = gpc / HW;
    const int  bpr  = gpc - bn * HW;
    const int  boh  = bpr / Wout;
    const int  bow  = bpr - boh * Wout;
    const int  ihb  = boh * stride - pad;
    const int  iwb  = bow * stride - pad;
    const float* xin = x + (size_t)bn * Cin * Hin * Win;
    _Float16* bdst  = &Bsw[gnc >> 4][krb + (gnc & 15)][0];

    v8f acc0 = {}, acc1 = {}, acc2 = {}, acc3 = {};

    for (int kt = 0; kt < Kg; kt += CONV_KT) {
        // ---- gather A (16 weights / thread, branchless) ----
        v8h wlo, whi;
        {
            const int kb = kt + kbsel;
            __builtin_prefetch(wrow + (kb + CONV_KT < Kg ? kb + CONV_KT : 0));
            #pragma unroll
            for (int j = 0; j < 8; ++j) {
                int k = kb + j;
                bool v = aocv && (k < Kg);
                float f = wrow[v ? k : 0];
                wlo[j] = (_Float16)(v ? f : 0.f);
            }
            #pragma unroll
            for (int j = 0; j < 8; ++j) {
                int k = kb + 8 + j;
                bool v = aocv && (k < Kg);
                float f = wrow[v ? k : 0];
                whi[j] = (_Float16)(v ? f : 0.f);
            }
        }
        // ---- gather B (16 im2col values / thread, branchless, incremental
        //      (c,kh,kw) decode: one pair of divides per K step) ----
        v16h bstage;
        {
            const int k0 = kt + krb;
            int c   = k0 / KHW;
            int rem = k0 - c * KHW;
            int kh  = rem / KW;
            int kw  = rem - kh * KW;
            #pragma unroll
            for (int j = 0; j < 16; ++j) {
                int  k  = k0 + j;
                int  ih = ihb + kh;
                int  iw = iwb + kw;
                bool valid = gpv && (k < Kg) &&
                             ((unsigned)ih < (unsigned)Hin) &&
                             ((unsigned)iw < (unsigned)Win);
                int  off = valid ? ((c * Hin + ih) * Win + iw) : 0;
                float f  = xin[off];
                bstage[j] = (_Float16)(valid ? f : 0.f);
                // carry-chain increment of (kw, kh, c)
                kw += 1;
                int ckw = (kw == KW) ? 1 : 0;
                kw = ckw ? 0 : kw;
                kh += ckw;
                int ckh = (kh == KH) ? 1 : 0;
                kh = ckh ? 0 : kh;
                c += ckh;
            }
        }
        // ---- store to swizzled LDS (contiguous b128 stores) ----
        *(v8h*)adst_lo = wlo;
        *(v8h*)adst_hi = whi;
        *(v16h*)bdst   = bstage;
        __syncthreads();

        // ---- 4 WMMAs per wave: one A fragment vs 4 B sub-tiles ----
        v16h a  = *(const v16h*)&Asw[wave][lane][0];
        v16h b0 = *(const v16h*)&Bsw[0][lane][0];
        v16h b1 = *(const v16h*)&Bsw[1][lane][0];
        v16h b2 = *(const v16h*)&Bsw[2][lane][0];
        v16h b3 = *(const v16h*)&Bsw[3][lane][0];
        acc0 = __builtin_amdgcn_wmma_f32_16x16x32_f16(false, a, false, b0, (short)0, acc0, false, false);
        acc1 = __builtin_amdgcn_wmma_f32_16x16x32_f16(false, a, false, b1, (short)0, acc1, false, false);
        acc2 = __builtin_amdgcn_wmma_f32_16x16x32_f16(false, a, false, b2, (short)0, acc2, false, false);
        acc3 = __builtin_amdgcn_wmma_f32_16x16x32_f16(false, a, false, b3, (short)0, acc3, false, false);
        __syncthreads();
    }

    // ---- epilogue: BN scale/bias (+residual) (+relu), scatter NCHW ----
    const int half = lane >> 4;
    const int ncol = lane & 15;
    #pragma unroll
    for (int t = 0; t < 4; ++t) {
        const v8f acc = (t == 0) ? acc0 : (t == 1) ? acc1 : (t == 2) ? acc2 : acc3;
        int gcol = npix_base + t * 16 + ncol;
        if (gcol < tot_pix) {
            int n  = gcol / HW;
            int pr = gcol - n * HW;
            #pragma unroll
            for (int r = 0; r < 8; ++r) {
                int oc = oc_base + wave * 16 + half * 8 + r;
                if (oc < Cout) {
                    float v = acc[r] * gamma[oc] + beta[oc];
                    int oidx = (n * Cout + oc) * HW + pr;
                    if (residual) v += residual[oidx];
                    if (relu_flag) v = v > 0.f ? v : 0.f;
                    out[oidx] = v;
                }
            }
        }
    }
}

// ---------------------------------------------------------------------------
// attention gate: out[n,c,p] = img[n,c,p] * softplus(sum_c aw[c]*img[n,c,p]+ab)
// ---------------------------------------------------------------------------
__global__ void attn_gate_kernel(const float* __restrict__ img,
                                 const float* __restrict__ aw,
                                 const float* __restrict__ ab,
                                 float* __restrict__ out, int N, int HW)
{
    int i = blockIdx.x * blockDim.x + threadIdx.x;
    int tot = N * HW;
    if (i >= tot) return;
    int n = i / HW, p = i - n * HW;
    const float* base = img + (size_t)n * 3 * HW + p;
    float s = aw[0] * base[0] + aw[1] * base[HW] + aw[2] * base[2 * HW] + ab[0];
    float sp = (s > 20.f) ? s : log1pf(expf(s));
    float* obase = out + (size_t)n * 3 * HW + p;
    obase[0]      = base[0]      * sp;
    obase[HW]     = base[HW]     * sp;
    obase[2 * HW] = base[2 * HW] * sp;
}

// 3x3 stride-2 pad-1 maxpool
__global__ void maxpool_kernel(const float* __restrict__ x, float* __restrict__ out,
                               int N, int C, int Hin, int Win, int Hout, int Wout)
{
    int i = blockIdx.x * blockDim.x + threadIdx.x;
    int tot = N * C * Hout * Wout;
    if (i >= tot) return;
    int ow = i % Wout; int t = i / Wout;
    int oh = t % Hout; t /= Hout;
    int c = t % C;     int n = t / C;
    float m = -3.402823466e38f;
    const float* base = x + ((size_t)(n * C + c)) * Hin * Win;
    #pragma unroll
    for (int kh = 0; kh < 3; ++kh) {
        int ih = oh * 2 - 1 + kh;
        if (ih < 0 || ih >= Hin) continue;
        #pragma unroll
        for (int kw = 0; kw < 3; ++kw) {
            int iw = ow * 2 - 1 + kw;
            if (iw < 0 || iw >= Win) continue;
            float v = base[ih * Win + iw];
            m = v > m ? v : m;
        }
    }
    out[i] = m;
}

// global average pool over HxW
__global__ void avgpool_kernel(const float* __restrict__ x, float* __restrict__ out,
                               int N, int C, int HW)
{
    int i = blockIdx.x * blockDim.x + threadIdx.x;
    if (i >= N * C) return;
    const float* base = x + (size_t)i * HW;
    float s = 0.f;
    for (int p = 0; p < HW; ++p) s += base[p];
    out[i] = s / (float)HW;
}

// ---------------------------------------------------------------------------
// Whole recurrence in one block: LSTM encoder (19 steps) + decoder (pred_len)
// ---------------------------------------------------------------------------
__device__ inline float sigf(float x) { return 1.f / (1.f + expf(-x)); }

__global__ __launch_bounds__(256)
void recurrent_kernel(const float* __restrict__ obs_pos,
                      const float* __restrict__ obs_pos_step,
                      const float* __restrict__ disp_W, const float* __restrict__ disp_b,
                      const float* __restrict__ Wi, const float* __restrict__ Wh,
                      const float* __restrict__ bi, const float* __restrict__ bh,
                      const float* __restrict__ h0,
                      const float* __restrict__ pred_W, const float* __restrict__ pred_b,
                      const int* __restrict__ pred_len_p,
                      const float* __restrict__ emb,   // [8][2048]
                      float* __restrict__ out)          // [8][pred_len][2]
{
    const int NB = 8, HID = 128, ED = 64, G = 512, CNN = 2048, OBS = 20;
    __shared__ float h[NB][HID], c[NB][HID], e[NB][ED], gates[NB][G];
    __shared__ float stepv[NB][2], pos[NB][2];
    const int tid = threadIdx.x;

    for (int i = tid; i < NB * HID; i += 256) {
        int n = i / HID, k = i - n * HID;
        h[n][k] = h0[k];
        c[n][k] = h0[k];
    }
    if (tid < NB * 2) {
        int n = tid >> 1, d = tid & 1;
        pos[n][d] = obs_pos[(n * OBS + (OBS - 1)) * 2 + d];
    }
    __syncthreads();

    // ---- encoder over obs_pos_step[:, 1:20] ----
    for (int t = 1; t < OBS; ++t) {
        for (int i = tid; i < NB * ED; i += 256) {
            int n = i / ED, k = i - n * ED;
            float s0 = obs_pos_step[(n * OBS + t) * 2 + 0];
            float s1 = obs_pos_step[(n * OBS + t) * 2 + 1];
            float v = s0 * disp_W[k] + s1 * disp_W[ED + k] + disp_b[k];
            e[n][k] = v > 0.f ? v : 0.f;
        }
        __syncthreads();
        for (int i = tid; i < NB * G; i += 256) {
            int n = i / G, g = i - n * G;
            float a = bi[g] + bh[g];
            for (int k = 0; k < ED; ++k)  a += e[n][k] * Wi[k * G + g];
            for (int k = 0; k < HID; ++k) a += h[n][k] * Wh[k * G + g];
            gates[n][g] = a;
        }
        __syncthreads();
        for (int i = tid; i < NB * HID; i += 256) {
            int n = i / HID, k = i - n * HID;
            float ig = gates[n][k], fg = gates[n][HID + k];
            float gg = gates[n][2 * HID + k], og = gates[n][3 * HID + k];
            float cn = sigf(fg) * c[n][k] + sigf(ig) * tanhf(gg);
            c[n][k] = cn;
            h[n][k] = sigf(og) * tanhf(cn);
        }
        __syncthreads();
    }

    // ---- decoder ----
    const int PL = *pred_len_p;
    for (int t = 0; t < PL; ++t) {
        if (tid < NB * 2) {
            int n = tid >> 1, d = tid & 1;
            float a = pred_b[d];
            for (int k = 0; k < HID; ++k) a += h[n][k] * pred_W[k * 2 + d];
            const float* eb = emb + n * CNN;
            for (int k = 0; k < CNN; ++k) a += eb[k] * pred_W[(HID + k) * 2 + d];
            float p = pos[n][d] + a;
            pos[n][d] = p;
            stepv[n][d] = a;
            out[(n * PL + t) * 2 + d] = p;
        }
        __syncthreads();
        for (int i = tid; i < NB * ED; i += 256) {
            int n = i / ED, k = i - n * ED;
            float v = stepv[n][0] * disp_W[k] + stepv[n][1] * disp_W[ED + k] + disp_b[k];
            e[n][k] = v > 0.f ? v : 0.f;
        }
        __syncthreads();
        for (int i = tid; i < NB * G; i += 256) {
            int n = i / G, g = i - n * G;
            float a = bi[g] + bh[g];
            for (int k = 0; k < ED; ++k)  a += e[n][k] * Wi[k * G + g];
            for (int k = 0; k < HID; ++k) a += h[n][k] * Wh[k * G + g];
            gates[n][g] = a;
        }
        __syncthreads();
        for (int i = tid; i < NB * HID; i += 256) {
            int n = i / HID, k = i - n * HID;
            float ig = gates[n][k], fg = gates[n][HID + k];
            float gg = gates[n][2 * HID + k], og = gates[n][3 * HID + k];
            float cn = sigf(fg) * c[n][k] + sigf(ig) * tanhf(gg);
            c[n][k] = cn;
            h[n][k] = sigf(og) * tanhf(cn);
        }
        __syncthreads();
    }
}

// ---------------------------------------------------------------------------
// Host-side orchestration
// ---------------------------------------------------------------------------
struct Blk {
    const float *w1, *g1, *b1, *w2, *g2, *b2, *w3, *g3, *b3, *wd, *gd, *bd;
};

extern "C" void kernel_launch(void* const* d_in, const int* in_sizes, int n_in,
                              void* d_out, int out_size, void* d_ws, size_t ws_size,
                              hipStream_t stream)
{
    (void)n_in; (void)out_size;
    auto F = [&](int i) { return (const float*)d_in[i]; };

    const float* img          = F(0);
    const float* obs_pos      = F(1);
    const float* obs_pos_step = F(2);
    const float* attn_w       = F(3);
    const float* attn_b       = F(4);

    static const int NST[4]   = {3, 4, 6, 3};
    static const int CCH[4]   = {64, 128, 256, 512};
    static const int SSTG[4]  = {1, 2, 2, 2};

    const float *stem_w, *stem_g, *stem_b;
    Blk blk[16];
    int p = 5;
    // Detect pytree flattening order of cnn_params:
    //   insertion order -> first entry is stem_w (9408 elems)
    //   sorted keys     -> 'blocks' first, first entry is blocks[0].b1 (64 elems)
    bool insertion = (in_sizes[5] == 64 * 3 * 7 * 7);
    if (insertion) {
        stem_w = F(p++); stem_g = F(p++); stem_b = F(p++);
        int k = 0;
        for (int s = 0; s < 4; ++s)
            for (int j = 0; j < NST[s]; ++j, ++k) {
                Blk& B = blk[k];
                B.w1 = F(p++); B.g1 = F(p++); B.b1 = F(p++);
                B.w2 = F(p++); B.g2 = F(p++); B.b2 = F(p++);
                B.w3 = F(p++); B.g3 = F(p++); B.b3 = F(p++);
                if (j == 0) { B.wd = F(p++); B.gd = F(p++); B.bd = F(p++); }
                else        { B.wd = nullptr; B.gd = nullptr; B.bd = nullptr; }
            }
    } else {
        int k = 0;
        for (int s = 0; s < 4; ++s)
            for (int j = 0; j < NST[s]; ++j, ++k) {
                Blk& B = blk[k];
                bool dn = (j == 0);
                B.b1 = F(p++); B.b2 = F(p++); B.b3 = F(p++);
                B.bd = dn ? F(p++) : nullptr;
                B.g1 = F(p++); B.g2 = F(p++); B.g3 = F(p++);
                B.gd = dn ? F(p++) : nullptr;
                B.w1 = F(p++); B.w2 = F(p++); B.w3 = F(p++);
                B.wd = dn ? F(p++) : nullptr;
            }
        stem_b = F(p++); stem_g = F(p++); stem_w = F(p++);
    }
    const float* disp_W   = F(p++);
    const float* disp_b   = F(p++);
    const float* Wi       = F(p++);
    const float* Wh       = F(p++);
    const float* bi       = F(p++);
    const float* bh       = F(p++);
    const float* h0       = F(p++);
    const float* pred_W   = F(p++);
    const float* pred_b   = F(p++);
    const int*   pred_len = (const int*)d_in[p++];

    // workspace layout: 4 rotating activation buffers + embedding
    const size_t BUF = (size_t)8 * 256 * 56 * 56;   // 6,422,528 floats (== stem out size)
    float* bufs[4];
    bufs[0] = (float*)d_ws;
    bufs[1] = bufs[0] + BUF;
    bufs[2] = bufs[1] + BUF;
    bufs[3] = bufs[2] + BUF;
    float* emb = bufs[3] + BUF;                      // 8*2048
    (void)ws_size;  // requires ~103 MB of workspace

    const int N = 8;
    auto conv = [&](const float* x, const float* w, const float* g, const float* b,
                    const float* res, float* o, int Cin, int Hin, int Win,
                    int Cout, int K, int stride, int pad, int Hout, int Wout, int relu) {
        int totpix = N * Hout * Wout;
        dim3 grid((totpix + CONV_NT - 1) / CONV_NT, (Cout + CONV_MT - 1) / CONV_MT);
        conv_bn_wmma_kernel<<<grid, 128, 0, stream>>>(
            x, w, g, b, res, o, N, Cin, Hin, Win, Cout, K, K, stride, pad, Hout, Wout, relu);
    };

    // 1) attention gate -> bufs[1]
    {
        int tot = N * 224 * 224;
        attn_gate_kernel<<<(tot + 255) / 256, 256, 0, stream>>>(
            img, attn_w, attn_b, bufs[1], N, 224 * 224);
    }
    // 2) stem conv 7x7/s2/p3 + bn + relu -> bufs[0]
    conv(bufs[1], stem_w, stem_g, stem_b, nullptr, bufs[0],
         3, 224, 224, 64, 7, 2, 3, 112, 112, 1);
    // 3) maxpool 3x3/s2/p1 -> bufs[1]
    {
        int tot = N * 64 * 56 * 56;
        maxpool_kernel<<<(tot + 255) / 256, 256, 0, stream>>>(
            bufs[0], bufs[1], N, 64, 112, 112, 56, 56);
    }

    // 4) bottleneck stack
    int icur = 1;
    int Ccur = 64, Hcur = 56;
    int k = 0;
    for (int s = 0; s < 4; ++s) {
        for (int j = 0; j < NST[s]; ++j, ++k) {
            const Blk& B   = blk[k];
            int cch        = CCH[s];
            int outc       = 4 * cch;
            int stride     = (j == 0) ? SSTG[s] : 1;
            int H2         = Hcur / stride;
            int o0 = -1, o1 = -1, o2 = -1;
            for (int q = 0, f = 0; q < 4; ++q)
                if (q != icur) { (f == 0 ? o0 : (f == 1 ? o1 : o2)) = q; ++f; }

            // conv1 1x1 + bn + relu
            conv(bufs[icur], B.w1, B.g1, B.b1, nullptr, bufs[o0],
                 Ccur, Hcur, Hcur, cch, 1, 1, 0, Hcur, Hcur, 1);
            // conv2 3x3/s + bn + relu
            conv(bufs[o0], B.w2, B.g2, B.b2, nullptr, bufs[o1],
                 cch, Hcur, Hcur, cch, 3, stride, 1, H2, H2, 1);
            // identity path
            const float* idn = bufs[icur];
            if (B.wd) {
                conv(bufs[icur], B.wd, B.gd, B.bd, nullptr, bufs[o0],
                     Ccur, Hcur, Hcur, outc, 1, stride, 0, H2, H2, 0);
                idn = bufs[o0];
            }
            // conv3 1x1 + bn + residual + relu
            conv(bufs[o1], B.w3, B.g3, B.b3, idn, bufs[o2],
                 cch, H2, H2, outc, 1, 1, 0, H2, H2, 1);

            icur = o2;
            Ccur = outc;
            Hcur = H2;
        }
    }

    // 5) global average pool -> emb (8 x 2048)
    {
        int tot = N * Ccur;
        avgpool_kernel<<<(tot + 255) / 256, 256, 0, stream>>>(
            bufs[icur], emb, N, Ccur, Hcur * Hcur);
    }

    // 6) LSTM encoder + autoregressive decoder -> d_out (8 x 30 x 2)
    recurrent_kernel<<<1, 256, 0, stream>>>(
        obs_pos, obs_pos_step, disp_W, disp_b, Wi, Wh, bi, bh, h0,
        pred_W, pred_b, pred_len, emb, (float*)d_out);
}
